// LocalStridedBlockSparseAttn_13443247636952
// MI455X (gfx1250) — matrix operand order repaired and checked by reading, not compiled, not run on hardware
//
#include <hip/hip_runtime.h>
#include <math.h>

typedef __attribute__((ext_vector_type(16))) _Float16 v16h;
typedef __attribute__((ext_vector_type(8)))  _Float16 v8h;
typedef __attribute__((ext_vector_type(8)))  float    v8f;

#define HQ 32
#define HK 8
#define GQA 4
#define HD 128
#define BLK 64
#define LOCAL_BLOCKS 16
#define VERT 8

// LDS row paddings keep 16-byte alignment of every fragment chunk and rotate banks
// (row strides: 136, 72, 72 halves = 272, 144, 144 bytes, all multiples of 16)
#define KPAD 8
#define VPAD 8
#define PPAD 8

__device__ __forceinline__ v16h ld_frag16(const _Float16* p0, const _Float16* p1) {
  union { v16h v; v8h h[2]; } u;
  u.h[0] = *(const v8h*)p0;
  u.h[1] = *(const v8h*)p1;
  return u.v;
}

// ---- DPP-based 16-lane butterfly reductions (no LDS round-trips) ----
template <int CTRL>
__device__ __forceinline__ float dpp_bcast(float x) {
  int xi = __builtin_bit_cast(int, x);
  int r  = __builtin_amdgcn_update_dpp(xi, xi, CTRL, 0xF, 0xF, true);
  return __builtin_bit_cast(float, r);
}
__device__ __forceinline__ float red_max16(float t) {
  t = fmaxf(t, dpp_bcast<0xB1>(t));   // quad_perm [1,0,3,2]  : xor 1
  t = fmaxf(t, dpp_bcast<0x4E>(t));   // quad_perm [2,3,0,1]  : xor 2
  t = fmaxf(t, dpp_bcast<0x141>(t));  // row_half_mirror      : completes xor 4
  t = fmaxf(t, dpp_bcast<0x140>(t));  // row_mirror           : completes xor 8
  return t;
}
__device__ __forceinline__ float red_sum16(float t) {
  t += dpp_bcast<0xB1>(t);
  t += dpp_bcast<0x4E>(t);
  t += dpp_bcast<0x141>(t);
  t += dpp_bcast<0x140>(t);
  return t;
}

__global__ __launch_bounds__(128) void sparse_attn_kernel(
    const float* __restrict__ q,
    const float* __restrict__ k,
    const float* __restrict__ v,
    float* __restrict__ out,
    int L, int nqb)
{
  __shared__ __align__(16) _Float16 sK[BLK][HD + KPAD];     // 64 x 136 (row-major K block)
  __shared__ __align__(16) _Float16 sVT[HD][BLK + VPAD];    // 128 x 72 (V transposed)
  __shared__ __align__(16) _Float16 sP[4][16][BLK + PPAD];  // per-wave P tile, 16 x 72

  const int tid  = threadIdx.x;
  const int wave = tid >> 5;
  const int lane = tid & 31;
  const int ln   = lane & 15;   // column / M index within fragment
  const int g    = lane >> 4;   // lane half-group

  int bid = blockIdx.x;
  const int qb = bid % nqb; bid /= nqb;
  const int hq = bid % HQ;  bid /= HQ;
  const int b  = bid;
  const int hk = hq / GQA;

  const int   seq0  = b * L;
  const float scale = rsqrtf((float)HD);

  // ---- Q strip (16 rows) loaded directly into WMMA A-fragment layout, pre-scaled ----
  const int qrow0 = qb * BLK + wave * 16;
  const float* qrow = q + ((size_t)(seq0 + qrow0 + ln) * HQ + hq) * HD;
  v16h qa[4];
  #pragma unroll
  for (int s = 0; s < 4; ++s) {
    union { v16h v; _Float16 e[16]; } u;
    const int k0 = s * 32 + g * 8;
    const float4 f0 = *(const float4*)(qrow + k0);
    const float4 f1 = *(const float4*)(qrow + k0 + 4);
    const float4 f2 = *(const float4*)(qrow + k0 + 16);
    const float4 f3 = *(const float4*)(qrow + k0 + 20);
    u.e[0]  = (_Float16)(f0.x * scale); u.e[1]  = (_Float16)(f0.y * scale);
    u.e[2]  = (_Float16)(f0.z * scale); u.e[3]  = (_Float16)(f0.w * scale);
    u.e[4]  = (_Float16)(f1.x * scale); u.e[5]  = (_Float16)(f1.y * scale);
    u.e[6]  = (_Float16)(f1.z * scale); u.e[7]  = (_Float16)(f1.w * scale);
    u.e[8]  = (_Float16)(f2.x * scale); u.e[9]  = (_Float16)(f2.y * scale);
    u.e[10] = (_Float16)(f2.z * scale); u.e[11] = (_Float16)(f2.w * scale);
    u.e[12] = (_Float16)(f3.x * scale); u.e[13] = (_Float16)(f3.y * scale);
    u.e[14] = (_Float16)(f3.z * scale); u.e[15] = (_Float16)(f3.w * scale);
    qa[s] = u.v;
  }

  v8f o[8];
  #pragma unroll
  for (int t = 0; t < 8; ++t) o[t] = (v8f)0.0f;
  float m8[8], l8[8];
  #pragma unroll
  for (int i = 0; i < 8; ++i) { m8[i] = -3.0e38f; l8[i] = 0.0f; }

  for (int kb = 0; kb <= qb; ++kb) {
    // block-sparse mask: local window OR head-dependent vertical stride (uniform per workgroup)
    const bool keep = ((qb - kb) < LOCAL_BLOCKS) || (((kb + hq + 1) % VERT) == 0);
    if (!keep) continue;

    // ---- stage K (row-major) and V (transposed) into LDS as f16, float4-vectorized ----
    {
      const float* kbase = k + ((size_t)(seq0 + kb * BLK) * HK + hk) * HD;
      const float* vbase = v + ((size_t)(seq0 + kb * BLK) * HK + hk) * HD;
      #pragma unroll
      for (int it = 0; it < (BLK * HD / 4) / 128; ++it) {
        const int idx = it * 128 + tid;
        const int r  = idx >> 5;          // row within block (128/4 = 32 float4 per row)
        const int c4 = idx & 31;          // float4 index within row
        const size_t go = (size_t)r * HK * HD + c4 * 4;
        const float4 kf = *(const float4*)(kbase + go);
        const float4 vf = *(const float4*)(vbase + go);
        union { _Float16 e[4]; unsigned long long u; } pk;
        pk.e[0] = (_Float16)kf.x; pk.e[1] = (_Float16)kf.y;
        pk.e[2] = (_Float16)kf.z; pk.e[3] = (_Float16)kf.w;
        *(unsigned long long*)&sK[r][c4 * 4] = pk.u;       // 8-byte aligned store
        sVT[c4 * 4 + 0][r] = (_Float16)vf.x;
        sVT[c4 * 4 + 1][r] = (_Float16)vf.y;
        sVT[c4 * 4 + 2][r] = (_Float16)vf.z;
        sVT[c4 * 4 + 3][r] = (_Float16)vf.w;
      }
    }
    __syncthreads();

    // ---- S = Q K^T : batch 4 B-fragment loads, then 4 WMMAs (one wait per 4 wmma) ----
    v8f s4[4];
    #pragma unroll
    for (int tn = 0; tn < 4; ++tn) s4[tn] = (v8f)0.0f;
    #pragma unroll
    for (int ss = 0; ss < 4; ++ss) {
      v16h bf[4];
      const int kb0 = ss * 32 + g * 8;
      #pragma unroll
      for (int tn = 0; tn < 4; ++tn) {
        const _Float16* krow = &sK[tn * 16 + ln][0];
        bf[tn] = ld_frag16(krow + kb0, krow + kb0 + 16);
      }
      #pragma unroll
      for (int tn = 0; tn < 4; ++tn)
        s4[tn] = __builtin_amdgcn_wmma_f32_16x16x32_f16(false, qa[ss], false, bf[tn],
                                                        (short)0, s4[tn], false, false);
    }

    // token-level causal mask on the diagonal block
    if (kb == qb) {
      #pragma unroll
      for (int tn = 0; tn < 4; ++tn) {
        const int colg = tn * 16 + ln;
        #pragma unroll
        for (int vv = 0; vv < 8; ++vv) {
          const int rowg = wave * 16 + g * 8 + vv;
          if (colg > rowg) s4[tn][vv] = -3.0e38f;
        }
      }
    }

    // ---- online softmax (row stats replicated over each 16-lane group, DPP reduces) ----
    float mnew[8], alpha[8];
    #pragma unroll
    for (int vv = 0; vv < 8; ++vv) {
      float t = fmaxf(fmaxf(s4[0][vv], s4[1][vv]), fmaxf(s4[2][vv], s4[3][vv]));
      t = red_max16(t);
      mnew[vv]  = fmaxf(m8[vv], t);
      alpha[vv] = __expf(m8[vv] - mnew[vv]);
      m8[vv]    = mnew[vv];
    }

    // P = exp(S - m), row sums, stash P (f16) in this wave's LDS tile
    float rs[8];
    #pragma unroll
    for (int vv = 0; vv < 8; ++vv) rs[vv] = 0.0f;
    #pragma unroll
    for (int tn = 0; tn < 4; ++tn) {
      #pragma unroll
      for (int vv = 0; vv < 8; ++vv) {
        const float p = __expf(s4[tn][vv] - mnew[vv]);
        rs[vv] += p;
        sP[wave][g * 8 + vv][tn * 16 + ln] = (_Float16)p;
      }
    }
    #pragma unroll
    for (int vv = 0; vv < 8; ++vv)
      l8[vv] = l8[vv] * alpha[vv] + red_sum16(rs[vv]);

    // rescale accumulators
    #pragma unroll
    for (int to = 0; to < 8; ++to)
      #pragma unroll
      for (int vv = 0; vv < 8; ++vv) o[to][vv] *= alpha[vv];

    // ---- O += P V : A-fragments shared across all 8 col-tiles; 4 loads then 4 WMMAs ----
    const _Float16* prow = &sP[wave][ln][0];
    v16h pa[2];
    #pragma unroll
    for (int ss = 0; ss < 2; ++ss) {
      const int a0 = ss * 32 + g * 8;
      pa[ss] = ld_frag16(prow + a0, prow + a0 + 16);
    }
    #pragma unroll
    for (int ss = 0; ss < 2; ++ss) {
      const int b0 = ss * 32 + g * 8;
      #pragma unroll
      for (int tg = 0; tg < 2; ++tg) {
        v16h bf[4];
        #pragma unroll
        for (int j = 0; j < 4; ++j) {
          const _Float16* vrow = &sVT[(tg * 4 + j) * 16 + ln][0];
          bf[j] = ld_frag16(vrow + b0, vrow + b0 + 16);
        }
        #pragma unroll
        for (int j = 0; j < 4; ++j)
          o[tg * 4 + j] = __builtin_amdgcn_wmma_f32_16x16x32_f16(false, pa[ss], false, bf[j],
                                                                 (short)0, o[tg * 4 + j],
                                                                 false, false);
      }
    }
    __syncthreads();  // protect sK/sVT before the next block overwrites them
  }

  // ---- epilogue: multiply by 1/row-sum, write f32 ----
  float inv[8];
  #pragma unroll
  for (int vv = 0; vv < 8; ++vv)
    inv[vv] = (l8[vv] > 0.0f) ? __frcp_rn(l8[vv]) : 0.0f;
  #pragma unroll
  for (int to = 0; to < 8; ++to) {
    const int d = to * 16 + ln;
    #pragma unroll
    for (int vv = 0; vv < 8; ++vv) {
      const int row = qrow0 + g * 8 + vv;
      out[((size_t)(seq0 + row) * HQ + hq) * HD + d] = o[to][vv] * inv[vv];
    }
  }
}

extern "C" void kernel_launch(void* const* d_in, const int* in_sizes, int n_in,
                              void* d_out, int out_size, void* d_ws, size_t ws_size,
                              hipStream_t stream) {
  (void)n_in; (void)d_ws; (void)ws_size; (void)out_size;
  const float* q = (const float*)d_in[0];
  const float* k = (const float*)d_in[1];
  const float* v = (const float*)d_in[2];
  // cu_seqlens_k describes equal-length sequences per the reference (L = T/B);
  // derive geometry host-side from sizes.
  const int T   = in_sizes[0] / (HQ * HD);
  const int B   = in_sizes[3] - 1;
  const int L   = T / B;
  const int nqb = L / BLK;
  dim3 grid((unsigned)(B * HQ * nqb));
  sparse_attn_kernel<<<grid, 128, 0, stream>>>(q, k, v, (float*)d_out, L, nqb);
}
